// SimpleModel_8418135900727
// MI455X (gfx1250) — matrix operand order repaired and checked by reading, not compiled, not run on hardware
//
#include <hip/hip_runtime.h>

// ---------------------------------------------------------------------------
// Fused 2-bit-quantized MLP for gfx1250 (MI455X), wave32 + v_wmma_f32_16x16x32_f16
//   out = (relu(x @ Wq1^T + b1)) @ Wq2^T + b2,  Wq = scale * clip(round(W/scale),-2,1)
// Quantized codes are exact in f16; scales folded into f32 epilogues.
// Dual-tile accumulator chains; layer-2 A-fragments register-resident.
// ---------------------------------------------------------------------------

typedef __attribute__((ext_vector_type(16))) _Float16 v16h;
typedef __attribute__((ext_vector_type(8)))  _Float16 v8h;
typedef __attribute__((ext_vector_type(8)))  float    v8f;
typedef __attribute__((ext_vector_type(4)))  float    v4f;

#define D_IN   256
#define D_HID  512
#define ALPHA  1.47f
#define WAVES  8

// LDS: per-wave h buffer, 16 rows x (512 + 16 pad) halves; rows 16B aligned
#define HROW   (D_HID + 16)              // 528 halves -> 1056 B row stride
#define HBUF_H (16 * HROW)               // 16896 B per wave
#define SMEM_BYTES (WAVES * HBUF_H * 2)  // 135168 B per workgroup (< 320KB WGP)

// ---------------- scale = alpha * mean|W| + 1e-8 (one block per tensor) -----
__global__ void scale_kernel(const float* __restrict__ W1,
                             const float* __restrict__ W2,
                             float* __restrict__ scales) {
    __shared__ float red[256];
    const float* W = (blockIdx.x == 0) ? W1 : W2;
    const int N = D_IN * D_HID;  // 131072 for both tensors
    float s = 0.f;
    for (int i = threadIdx.x; i < N; i += 256) s += fabsf(W[i]);
    red[threadIdx.x] = s;
    __syncthreads();
    for (int off = 128; off > 0; off >>= 1) {
        if ((int)threadIdx.x < off) red[threadIdx.x] += red[threadIdx.x + off];
        __syncthreads();
    }
    if (threadIdx.x == 0)
        scales[blockIdx.x] = ALPHA * (red[0] / (float)N) + 1e-8f;
}

// ---------------- quantize codes to f16 in {-2,-1,0,1} ----------------------
__global__ void quant_kernel(const float* __restrict__ W1,
                             const float* __restrict__ W2,
                             const float* __restrict__ scales,
                             _Float16* __restrict__ Wq1,
                             _Float16* __restrict__ Wq2) {
    int idx = blockIdx.x * blockDim.x + threadIdx.x;
    const int N = D_IN * D_HID;
    if (idx < N) {
        float q = rintf(W1[idx] / scales[0]);
        q = fminf(fmaxf(q, -2.f), 1.f);
        Wq1[idx] = (_Float16)q;
    } else {
        int j = idx - N;
        float q = rintf(W2[j] / scales[1]);
        q = fminf(fmaxf(q, -2.f), 1.f);
        Wq2[j] = (_Float16)q;
    }
}

// ---- B-fragment (32x16 f16): lane n=L%16, elems 0-7 -> k+0..7,
//      elems 8-15 -> k+16..23, with k already including (L>=16 ? 8 : 0)
__device__ __forceinline__ v16h load_fragB(const _Float16* __restrict__ p) {
    v8h lo = *(const v8h*)(p);
    v8h hi = *(const v8h*)(p + 16);
    v16h f;
#pragma unroll
    for (int i = 0; i < 8; ++i) { f[i] = lo[i]; f[i + 8] = hi[i]; }
    return f;
}

// ---- A-fragment from LDS (16x32 f16): lane m=L%16, elems 0-7 -> k+0..7,
//      elems 8-15 -> k+16..23, with k already including (L>=16 ? 8 : 0)
__device__ __forceinline__ v16h load_fragA_lds(const _Float16* __restrict__ p) {
    v8h lo = *(const v8h*)(p);
    v8h hi = *(const v8h*)(p + 16);
    v16h f;
#pragma unroll
    for (int i = 0; i < 8; ++i) { f[i] = lo[i]; f[i + 8] = hi[i]; }
    return f;
}

__device__ __forceinline__ v8f wmma_f16(const v16h& a, const v16h& b, const v8f& c) {
    return __builtin_amdgcn_wmma_f32_16x16x32_f16(
        false, a, false, b, (short)0, c, false, false);
}

// ---------------- fused 2-layer kernel --------------------------------------
__global__ void __launch_bounds__(WAVES * 32)
mlp_kernel(const float* __restrict__ x,
           const _Float16* __restrict__ Wq1, const float* __restrict__ b1,
           const _Float16* __restrict__ Wq2, const float* __restrict__ b2,
           const float* __restrict__ scales,
           float* __restrict__ out) {
    extern __shared__ _Float16 smem[];
    const int lane = threadIdx.x & 31;
    const int wave = threadIdx.x >> 5;
    _Float16* hbuf = smem + wave * HBUF_H;

    const int m     = lane & 15;           // A: row / B: col / D: col index
    const int hiH   = lane >> 4;           // 0 for lanes 0-15, 1 for 16-31
    const int koffA = hiH * 8;             // A-fragment K base offset
    const int koffB = hiH * 8;             // B-fragment K base offset
    const int moff  = hiH * 8;             // D-fragment row offset

    const long row0 = ((long)blockIdx.x * WAVES + wave) * 16;
    const float s1 = scales[0];
    const float s2 = scales[1];

    // ---- load + convert the wave's 16 x rows into 8 cached A-fragments -----
    v16h afrag[D_IN / 32];
    {
        const float* xrow = x + (row0 + m) * D_IN;
#pragma unroll
        for (int kt = 0; kt < D_IN / 32; ++kt) {
            const int k0 = kt * 32 + koffA;
            v4f f0 = *(const v4f*)(xrow + k0);
            v4f f1 = *(const v4f*)(xrow + k0 + 4);
            v4f f2 = *(const v4f*)(xrow + k0 + 16);
            v4f f3 = *(const v4f*)(xrow + k0 + 20);
            v16h a;
#pragma unroll
            for (int i = 0; i < 4; ++i) {
                a[i]      = (_Float16)f0[i];
                a[i + 4]  = (_Float16)f1[i];
                a[i + 8]  = (_Float16)f2[i];
                a[i + 12] = (_Float16)f3[i];
            }
            afrag[kt] = a;
        }
    }

    // ---- layer 1: h = relu(s1 * (x @ q1^T) + b1), two tiles per pass -------
    for (int nt = 0; nt < D_HID / 16; nt += 2) {
        const _Float16* w0 = Wq1 + (size_t)(nt * 16 + m) * D_IN + koffB;
        const _Float16* w1 = w0 + (size_t)16 * D_IN;
        // prefetch next pass's weight rows into cache (global_prefetch_b8)
        if (nt + 2 < D_HID / 16)
            __builtin_prefetch(w0 + (size_t)32 * D_IN, 0, 3);
        v8f acc0 = {};
        v8f acc1 = {};
#pragma unroll
        for (int kt = 0; kt < D_IN / 32; ++kt) {
            v16h bf0 = load_fragB(w0 + kt * 32);
            v16h bf1 = load_fragB(w1 + kt * 32);
            acc0 = wmma_f16(afrag[kt], bf0, acc0);
            acc1 = wmma_f16(afrag[kt], bf1, acc1);
        }
        const float bv0 = b1[nt * 16 + m];
        const float bv1 = b1[nt * 16 + 16 + m];
        _Float16* hc0 = hbuf + nt * 16 + m;        // column, row stride HROW
        _Float16* hc1 = hc0 + 16;
#pragma unroll
        for (int j = 0; j < 8; ++j) {
            float h0 = s1 * acc0[j] + bv0;
            float h1 = s1 * acc1[j] + bv1;
            h0 = h0 > 0.f ? h0 : 0.f;
            h1 = h1 > 0.f ? h1 : 0.f;
            hc0[(j + moff) * HROW] = (_Float16)h0;
            hc1[(j + moff) * HROW] = (_Float16)h1;
        }
    }

    __syncthreads();   // h tiles -> LDS visible for the A-layout reload

    // ---- reload h once as 16 register-resident A-fragments -----------------
    v16h hfrag[D_HID / 32];
    {
        const _Float16* hrow = hbuf + m * HROW + koffA;
#pragma unroll
        for (int kt = 0; kt < D_HID / 32; ++kt)
            hfrag[kt] = load_fragA_lds(hrow + kt * 32);
    }

    // ---- layer 2: out = s2 * (h @ q2^T) + b2, two tiles per pass -----------
    for (int ct = 0; ct < D_IN / 16; ct += 2) {
        const _Float16* w0 = Wq2 + (size_t)(ct * 16 + m) * D_HID + koffB;
        const _Float16* w1 = w0 + (size_t)16 * D_HID;
        if (ct + 2 < D_IN / 16)
            __builtin_prefetch(w0 + (size_t)32 * D_HID, 0, 3);
        v8f acc0 = {};
        v8f acc1 = {};
#pragma unroll
        for (int kt = 0; kt < D_HID / 32; ++kt) {
            v16h bf0 = load_fragB(w0 + kt * 32);
            v16h bf1 = load_fragB(w1 + kt * 32);
            acc0 = wmma_f16(hfrag[kt], bf0, acc0);
            acc1 = wmma_f16(hfrag[kt], bf1, acc1);
        }
        const float bv0 = b2[ct * 16 + m];
        const float bv1 = b2[ct * 16 + 16 + m];
        float* o0 = out + row0 * D_IN + ct * 16 + m;
        float* o1 = o0 + 16;
#pragma unroll
        for (int j = 0; j < 8; ++j) {
            o0[(size_t)(j + moff) * D_IN] = s2 * acc0[j] + bv0;
            o1[(size_t)(j + moff) * D_IN] = s2 * acc1[j] + bv1;
        }
    }
}

// ---------------------------------------------------------------------------
extern "C" void kernel_launch(void* const* d_in, const int* in_sizes, int n_in,
                              void* d_out, int out_size, void* d_ws, size_t ws_size,
                              hipStream_t stream) {
    const float* x  = (const float*)d_in[0];
    const float* W1 = (const float*)d_in[1];
    const float* b1 = (const float*)d_in[2];
    const float* W2 = (const float*)d_in[3];
    const float* b2 = (const float*)d_in[4];
    float* out = (float*)d_out;

    // workspace: [0..1] scales (f32), then f16 code matrices (16B aligned)
    float*    scales = (float*)d_ws;
    _Float16* Wq1    = (_Float16*)((char*)d_ws + 64);
    _Float16* Wq2    = Wq1 + D_IN * D_HID;

    scale_kernel<<<2, 256, 0, stream>>>(W1, W2, scales);
    quant_kernel<<<(2 * D_IN * D_HID) / 256, 256, 0, stream>>>(
        W1, W2, scales, Wq1, Wq2);

    const int B = in_sizes[0] / D_IN;          // 262144
    const int blocks = B / (WAVES * 16);       // 2048
    hipFuncSetAttribute((const void*)mlp_kernel,
                        hipFuncAttributeMaxDynamicSharedMemorySize, SMEM_BYTES);
    mlp_kernel<<<blocks, WAVES * 32, SMEM_BYTES, stream>>>(
        x, Wq1, b1, Wq2, b2, scales, out);
}
